// GAT_83245056131910
// MI455X (gfx1250) — compile-verified, matrix operand order checked by reading, not run on hardware
//
#include <hip/hip_runtime.h>
#include <math.h>

typedef __attribute__((ext_vector_type(16))) _Float16 v16h;
typedef __attribute__((ext_vector_type(8)))  _Float16 v8h;
typedef __attribute__((ext_vector_type(8)))  float    v8f;
typedef __attribute__((ext_vector_type(4)))  float    v4f;

#define N_NODES 50000
#define E_RAW   800000
#define E_TOT   (E_RAW + N_NODES)   // self loops appended
#define F       256                 // feature width (layers 1-2)
#define H8      8
#define C32     32

// ---------- helpers ----------

// order-preserving float -> u32 key (monotonic), for atomic segment-max
__device__ __forceinline__ unsigned int f2key(float f) {
    unsigned int b = __float_as_uint(f);
    return (b & 0x80000000u) ? ~b : (b | 0x80000000u);
}
__device__ __forceinline__ float key2f(unsigned int k) {
    return __uint_as_float((k & 0x80000000u) ? (k & 0x7fffffffu) : ~k);
}

__device__ __forceinline__ void edge_sd(const int* __restrict__ ei, int e, int& s, int& d) {
    if (e < E_RAW) { s = ei[e]; d = ei[E_RAW + e]; }
    else           { s = d = e - E_RAW; }          // self loop
}

__device__ __forceinline__ float lrelu02(float v) { return v > 0.f ? v : 0.2f * v; }

// ---------- generic fill ----------
__global__ void fill_u32_k(unsigned int* p, unsigned int v, long n) {
    long i = (long)blockIdx.x * blockDim.x + threadIdx.x;
    if (i < n) p[i] = v;
}

// ---------- W [K,N] f32 row-major -> Wt [N,K] f16 ----------
__global__ void convert_wt_k(const float* __restrict__ W, _Float16* __restrict__ Wt) {
    int i = blockIdx.x * blockDim.x + threadIdx.x;   // 65536
    if (i >= F * F) return;
    int k = i >> 8, n = i & (F - 1);                 // coalesced read of W
    Wt[n * F + k] = (_Float16)W[i];
}

// ---------- WMMA GEMM:  Y[M,256] = X[M,256] @ W[256,256] ----------
// Wt is the f16 transposed weight [N][K]. Block = 128 threads = 4 waves.
// Each wave computes a 16x64 strip (4 accumulators), reusing the A fragment.
// grid = (M/16); the 4 waves cover all 256 output columns of 16 rows.
__global__ void __launch_bounds__(128) gemm_wmma_k(
    const float* __restrict__ X, const _Float16* __restrict__ Wt,
    float* __restrict__ Y) {
    const int lane = threadIdx.x & 31;
    const int wave = threadIdx.x >> 5;
    const int m0 = blockIdx.x * 16;
    const int n0 = wave * 64;

    const int mrow = m0 + (lane & 15);      // A row this lane feeds
    const int kb   = (lane < 16) ? 0 : 8;   // K sub-offset per lane half

    v8f c0 = {}, c1 = {}, c2 = {}, c3 = {};
    for (int kk = 0; kk < F; kk += 32) {
        // A fragment: rows of X, two aligned 8-float runs, cvt f32->f16
        const float* ap0 = X + (long)mrow * F + kk + kb;
        v4f a0 = *reinterpret_cast<const v4f*>(ap0);
        v4f a1 = *reinterpret_cast<const v4f*>(ap0 + 4);
        v4f a2 = *reinterpret_cast<const v4f*>(ap0 + 16);
        v4f a3 = *reinterpret_cast<const v4f*>(ap0 + 20);
        v16h a;
#pragma unroll
        for (int i = 0; i < 4; i++) {
            a[i]      = (_Float16)a0[i];
            a[4 + i]  = (_Float16)a1[i];
            a[8 + i]  = (_Float16)a2[i];
            a[12 + i] = (_Float16)a3[i];
        }
        // B fragments: Wt row-major [N][K] f16 -> two b128 loads per tile
        v8f* accs[4] = { &c0, &c1, &c2, &c3 };
#pragma unroll
        for (int t = 0; t < 4; t++) {
            const _Float16* bp = Wt + (long)(n0 + t * 16 + (lane & 15)) * F + kk + kb;
            v8h blo = *reinterpret_cast<const v8h*>(bp);
            v8h bhi = *reinterpret_cast<const v8h*>(bp + 16);
            v16h b;
#pragma unroll
            for (int i = 0; i < 8; i++) { b[i] = blo[i]; b[8 + i] = bhi[i]; }
            *accs[t] = __builtin_amdgcn_wmma_f32_16x16x32_f16(
                false, a, false, b, (short)0, *accs[t], false, false);
        }
    }

    const int row = m0 + (lane < 16 ? 0 : 8);
    v8f cc[4] = { c0, c1, c2, c3 };
#pragma unroll
    for (int t = 0; t < 4; t++) {
        const int col = n0 + t * 16 + (lane & 15);
#pragma unroll
        for (int r = 0; r < 8; r++)
            Y[(long)(row + r) * F + col] = cc[t][r];
    }
}

// ---------- attention coefficients: AS/AD[n,h] = h[n,h,:] . a_{src,dst}[h,:] ----------
__global__ void attn_coeff_k(const float* __restrict__ h,
                             const float* __restrict__ asrc, const float* __restrict__ adst,
                             float* __restrict__ AS, float* __restrict__ AD,
                             int N, int Hh, int C) {
    int tid = blockIdx.x * blockDim.x + threadIdx.x;
    if (tid >= N * Hh) return;
    int n = tid / Hh, hh = tid % Hh;
    const float* hp = h + (long)n * Hh * C + (long)hh * C;
    const float* s = asrc + hh * C;
    const float* d = adst + hh * C;
    float accs = 0.f, accd = 0.f;
    for (int c = 0; c < C; c++) { float v = hp[c]; accs += v * s[c]; accd += v * d[c]; }
    AS[tid] = accs; AD[tid] = accd;
}

// ---------- edge pass 1: segment max (ordered-key atomicMax) ----------
__global__ void edge_max_k(const int* __restrict__ ei, const float* __restrict__ AS,
                           const float* __restrict__ AD, unsigned int* __restrict__ MK, int Hh) {
    long tid = (long)blockIdx.x * blockDim.x + threadIdx.x;
    if (tid >= (long)E_TOT * Hh) return;
    int e = (int)(tid / Hh), hh = (int)(tid - (long)e * Hh);
    int s, d; edge_sd(ei, e, s, d);
    float ev = lrelu02(AS[s * Hh + hh] + AD[d * Hh + hh]);
    atomicMax(&MK[d * Hh + hh], f2key(ev));
}

// ---------- edge pass 2: segment sum of exp(e - max) ----------
__global__ void edge_sum_k(const int* __restrict__ ei, const float* __restrict__ AS,
                           const float* __restrict__ AD, const unsigned int* __restrict__ MK,
                           float* __restrict__ DEN, int Hh) {
    long tid = (long)blockIdx.x * blockDim.x + threadIdx.x;
    if (tid >= (long)E_TOT * Hh) return;
    int e = (int)(tid / Hh), hh = (int)(tid - (long)e * Hh);
    int s, d; edge_sd(ei, e, s, d);
    float ev = lrelu02(AS[s * Hh + hh] + AD[d * Hh + hh]);
    float ee = __expf(ev - key2f(MK[d * Hh + hh]));
    atomicAdd(&DEN[d * Hh + hh], ee);
}

// ---------- edge pass 3 (layers 1-2): scatter h[src]*alpha, one wave per edge ----------
__global__ void __launch_bounds__(256) edge_scatter8_k(
    const int* __restrict__ ei, const float* __restrict__ AS, const float* __restrict__ AD,
    const unsigned int* __restrict__ MK, const float* __restrict__ DEN,
    const float* __restrict__ hsrc, float* __restrict__ aggr) {
    const int lane = threadIdx.x & 31;
    long gw = ((long)blockIdx.x * blockDim.x + threadIdx.x) >> 5;
    if (gw >= E_TOT) return;
    int e = (int)gw;
    int s, d; edge_sd(ei, e, s, d);

    float alpha_l = 0.f;
    if (lane < H8) {
        float ev = lrelu02(AS[s * H8 + lane] + AD[d * H8 + lane]);
        float ee = __expf(ev - key2f(MK[d * H8 + lane]));
        alpha_l = ee / (DEN[d * H8 + lane] + 1e-16f);
    }
    const float* hp = hsrc + (long)s * F;
    float* op = aggr + (long)d * F;
#pragma unroll
    for (int j = 0; j < H8; j++) {
        float alpha = __shfl(alpha_l, j, 32);
        atomicAdd(&op[j * C32 + lane], hp[j * C32 + lane] * alpha);
    }
}

// ---------- bias + ELU ----------
__global__ void bias_elu_k(const float* __restrict__ x, const float* __restrict__ b,
                           float* __restrict__ y, long n) {
    long i = (long)blockIdx.x * blockDim.x + threadIdx.x;
    if (i >= n) return;
    float v = x[i] + b[i & (F - 1)];
    y[i] = v > 0.f ? v : (__expf(v) - 1.f);
}

// ---------- layer 3 projection + attention coeffs (OUT_DIM=2, heads=1) ----------
__global__ void layer3_proj_k(const float* __restrict__ X, const float* __restrict__ W3,
                              const float* __restrict__ as3, const float* __restrict__ ad3,
                              float* __restrict__ T3, float* __restrict__ AS, float* __restrict__ AD,
                              int N) {
    int n = blockIdx.x * blockDim.x + threadIdx.x;
    if (n >= N) return;
    const float* xp = X + (long)n * F;
    float a0 = 0.f, a1 = 0.f;
    for (int k = 0; k < F; k++) { float v = xp[k]; a0 += v * W3[2 * k]; a1 += v * W3[2 * k + 1]; }
    T3[2 * n] = a0; T3[2 * n + 1] = a1;
    AS[n] = a0 * as3[0] + a1 * as3[1];
    AD[n] = a0 * ad3[0] + a1 * ad3[1];
}

// ---------- layer 3 scatter (H=1, C=2), thread per edge ----------
__global__ void edge_scatter3_k(const int* __restrict__ ei, const float* __restrict__ AS,
                                const float* __restrict__ AD, const unsigned int* __restrict__ MK,
                                const float* __restrict__ DEN, const float* __restrict__ T3,
                                float* __restrict__ out) {
    int e = blockIdx.x * blockDim.x + threadIdx.x;
    if (e >= E_TOT) return;
    int s, d; edge_sd(ei, e, s, d);
    float ev = lrelu02(AS[s] + AD[d]);
    float ee = __expf(ev - key2f(MK[d]));
    float alpha = ee / (DEN[d] + 1e-16f);
    atomicAdd(&out[2 * d],     T3[2 * s] * alpha);
    atomicAdd(&out[2 * d + 1], T3[2 * s + 1] * alpha);
}

__global__ void add_b3_k(float* out, const float* __restrict__ b3, int n2) {
    int i = blockIdx.x * blockDim.x + threadIdx.x;
    if (i < n2) out[i] += b3[i & 1];
}

// ==================== launch ====================
static inline int cdiv(long a, int b) { return (int)((a + b - 1) / b); }

extern "C" void kernel_launch(void* const* d_in, const int* in_sizes, int n_in,
                              void* d_out, int out_size, void* d_ws, size_t ws_size,
                              hipStream_t stream) {
    (void)in_sizes; (void)n_in; (void)out_size; (void)ws_size;
    const float* x   = (const float*)d_in[0];
    const int*   ei  = (const int*)d_in[1];
    const float* W1  = (const float*)d_in[2];
    const float* as1 = (const float*)d_in[3];
    const float* ad1 = (const float*)d_in[4];
    const float* b1  = (const float*)d_in[5];
    const float* W2  = (const float*)d_in[6];
    const float* as2 = (const float*)d_in[7];
    const float* ad2 = (const float*)d_in[8];
    const float* b2  = (const float*)d_in[9];
    const float* W3  = (const float*)d_in[10];
    const float* as3 = (const float*)d_in[11];
    const float* ad3 = (const float*)d_in[12];
    const float* b3  = (const float*)d_in[13];
    float* out = (float*)d_out;

    // workspace layout (bytes)
    char* ws = (char*)d_ws;
    const long NF = (long)N_NODES * F;            // 12.8M floats
    float*        H    = (float*)(ws);
    float*        X2   = (float*)(ws + 4 * NF);
    float*        AGGR = (float*)(ws + 8 * NF);
    float*        AS   = (float*)(ws + 12 * NF);                      // N*8
    float*        AD   = AS + (long)N_NODES * H8;
    unsigned int* MK   = (unsigned int*)(AD + (long)N_NODES * H8);
    float*        DEN  = (float*)(MK + (long)N_NODES * H8);
    float*        T3   = DEN + (long)N_NODES * H8;                    // N*2
    float*        AS3  = T3 + 2L * N_NODES;
    float*        AD3  = AS3 + N_NODES;
    unsigned int* MK3  = (unsigned int*)(AD3 + N_NODES);
    float*        DEN3 = (float*)(MK3 + N_NODES);
    _Float16*     WT   = (_Float16*)(DEN3 + N_NODES);                 // 256*256 f16 (16B aligned)

    const long NH = (long)N_NODES * H8;
    const long EH = (long)E_TOT * H8;

    for (int layer = 0; layer < 2; layer++) {
        const float* Win  = layer == 0 ? W1 : W2;
        const float* asl  = layer == 0 ? as1 : as2;
        const float* adl  = layer == 0 ? ad1 : ad2;
        const float* bl   = layer == 0 ? b1 : b2;
        const float* Xin  = layer == 0 ? x : X2;

        convert_wt_k<<<cdiv(F * F, 256), 256, 0, stream>>>(Win, WT);
        gemm_wmma_k<<<N_NODES / 16, 128, 0, stream>>>(Xin, WT, H);
        attn_coeff_k<<<cdiv(NH, 256), 256, 0, stream>>>(H, asl, adl, AS, AD, N_NODES, H8, C32);
        fill_u32_k<<<cdiv(NH, 256), 256, 0, stream>>>(MK, 0u, NH);
        fill_u32_k<<<cdiv(NH, 256), 256, 0, stream>>>((unsigned int*)DEN, 0u, NH);
        fill_u32_k<<<cdiv(NF, 256), 256, 0, stream>>>((unsigned int*)AGGR, 0u, NF);
        edge_max_k<<<cdiv(EH, 256), 256, 0, stream>>>(ei, AS, AD, MK, H8);
        edge_sum_k<<<cdiv(EH, 256), 256, 0, stream>>>(ei, AS, AD, MK, DEN, H8);
        edge_scatter8_k<<<cdiv((long)E_TOT * 32, 256), 256, 0, stream>>>(ei, AS, AD, MK, DEN, H, AGGR);
        bias_elu_k<<<cdiv(NF, 256), 256, 0, stream>>>(AGGR, bl, X2, NF);
    }

    // layer 3: hidden -> 2, heads=1, concat=False (mean over 1 head = identity)
    layer3_proj_k<<<cdiv(N_NODES, 256), 256, 0, stream>>>(X2, W3, as3, ad3, T3, AS3, AD3, N_NODES);
    fill_u32_k<<<cdiv(N_NODES, 256), 256, 0, stream>>>(MK3, 0u, N_NODES);
    fill_u32_k<<<cdiv(N_NODES, 256), 256, 0, stream>>>((unsigned int*)DEN3, 0u, N_NODES);
    fill_u32_k<<<cdiv(2L * N_NODES, 256), 256, 0, stream>>>((unsigned int*)out, 0u, 2L * N_NODES);
    edge_max_k<<<cdiv((long)E_TOT, 256), 256, 0, stream>>>(ei, AS3, AD3, MK3, 1);
    edge_sum_k<<<cdiv((long)E_TOT, 256), 256, 0, stream>>>(ei, AS3, AD3, MK3, DEN3, 1);
    edge_scatter3_k<<<cdiv((long)E_TOT, 256), 256, 0, stream>>>(ei, AS3, AD3, MK3, DEN3, T3, out);
    add_b3_k<<<cdiv(2L * N_NODES, 256), 256, 0, stream>>>(out, b3, 2 * N_NODES);
}
